// W8DWLinear_18502719111673
// MI455X (gfx1250) — 8-bit matrix operand order repaired and checked by reading, compiled, not run on hardware
//
#include <hip/hip_runtime.h>
#include <stdint.h>

typedef int   i32x8 __attribute__((ext_vector_type(8)));
typedef int   i32x4 __attribute__((ext_vector_type(4)));
typedef float f32x4 __attribute__((ext_vector_type(4)));

#define K_OUT 16
#define QF    127.0f
#define I_DIM 2048
#define O_DIM 2048
#define B_DIM 4096

__global__ void colmax_kernel(const float* __restrict__ w, float* __restrict__ col_max) {
    int c = blockIdx.x * blockDim.x + threadIdx.x;
    float m = 0.0f;
    for (int o = 0; o < O_DIM; ++o)
        m = fmaxf(m, fabsf(w[(size_t)o * I_DIM + c]));
    col_max[c] = m;
}

__global__ void topk_kernel(const float* __restrict__ col_max, int* __restrict__ idx) {
    __shared__ float sm[I_DIM];
    __shared__ float rv[256];
    __shared__ int   ri[256];
    int tid = threadIdx.x;
    for (int i = tid; i < I_DIM; i += 256) sm[i] = col_max[i];
    __syncthreads();
    for (int k = 0; k < K_OUT; ++k) {
        float bv = -1.0f; int bi = 0;
        for (int i = tid; i < I_DIM; i += 256) {
            float v = sm[i];
            if (v > bv) { bv = v; bi = i; }
        }
        rv[tid] = bv; ri[tid] = bi;
        __syncthreads();
        for (int s = 128; s > 0; s >>= 1) {
            if (tid < s) {
                if (rv[tid + s] > rv[tid] ||
                    (rv[tid + s] == rv[tid] && ri[tid + s] < ri[tid])) {
                    rv[tid] = rv[tid + s]; ri[tid] = ri[tid + s];
                }
            }
            __syncthreads();
        }
        if (tid == 0) { idx[k] = ri[0]; sm[ri[0]] = -1.0f; }
        __syncthreads();
    }
}

__global__ __launch_bounds__(256)
void quant_rows_kernel(const float* __restrict__ in, const int* __restrict__ idx,
                       signed char* __restrict__ q, float* __restrict__ rmax,
                       float* __restrict__ unq) {
    __shared__ float red[256];
    __shared__ int   sidx[K_OUT];
    int tid = threadIdx.x;
    size_t row = blockIdx.x;
    if (tid < K_OUT) sidx[tid] = idx[tid];
    __syncthreads();

    const float* rp = in + row * I_DIM;
    float vals[I_DIM / 256];
    float m = 0.0f;
#pragma unroll
    for (int j = 0; j < I_DIM / 256; ++j) {
        int c = tid + j * 256;
        float v = rp[c];
        bool is_out = false;
#pragma unroll
        for (int t = 0; t < K_OUT; ++t) is_out |= (sidx[t] == c);
        v = is_out ? 0.0f : v;
        vals[j] = v;
        m = fmaxf(m, fabsf(v));
    }
    red[tid] = m;
    __syncthreads();
    for (int s = 128; s > 0; s >>= 1) {
        if (tid < s) red[tid] = fmaxf(red[tid], red[tid + s]);
        __syncthreads();
    }
    float scale = QF / red[0];
    signed char* qp = q + row * I_DIM;
#pragma unroll
    for (int j = 0; j < I_DIM / 256; ++j) {
        int c = tid + j * 256;
        qp[c] = (signed char)(int)(vals[j] * scale);
    }
    if (tid == 0) rmax[row] = red[0];
    if (tid < K_OUT) unq[row * K_OUT + tid] = rp[sidx[tid]];
}

__global__ __launch_bounds__(256)
void gemm_kernel(const signed char* __restrict__ x_q, const signed char* __restrict__ w_q,
                 const float* __restrict__ x_max, const float* __restrict__ w_max,
                 const float* __restrict__ x_unq, const float* __restrict__ wcol,
                 const float* __restrict__ bias, float* __restrict__ out) {
    const int lane = threadIdx.x & 31;
    const int wave = threadIdx.x >> 5;
    const int lm   = lane & 15;
    const int half = lane >> 4;
    const int m_base = blockIdx.y * 128 + wave * 16;
    const int n_base = blockIdx.x * 64;

    const signed char* pa  = x_q + (size_t)(m_base + lm) * I_DIM + 16 * half;
    const signed char* pb0 = w_q + (size_t)(n_base +  0 + lm) * I_DIM + 16 * half;
    const signed char* pb1 = w_q + (size_t)(n_base + 16 + lm) * I_DIM + 16 * half;
    const signed char* pb2 = w_q + (size_t)(n_base + 32 + lm) * I_DIM + 16 * half;
    const signed char* pb3 = w_q + (size_t)(n_base + 48 + lm) * I_DIM + 16 * half;

    i32x8 acc0 = {}, acc1 = {}, acc2 = {}, acc3 = {};

#pragma unroll 1
    for (int k = 0; k < I_DIM; k += 64) {
        union { i32x8 v; unsigned long long q[4]; } A;
        A.q[0] = *(const unsigned long long*)(pa + k +  0);
        A.q[1] = *(const unsigned long long*)(pa + k +  8);
        A.q[2] = *(const unsigned long long*)(pa + k + 32);
        A.q[3] = *(const unsigned long long*)(pa + k + 40);

        union { i32x8 v; i32x4 u[2]; } B0, B1, B2, B3;
        B0.u[0] = *(const i32x4*)(pb0 + k);
        B0.u[1] = *(const i32x4*)(pb0 + k + 32);
        B1.u[0] = *(const i32x4*)(pb1 + k);
        B1.u[1] = *(const i32x4*)(pb1 + k + 32);
        B2.u[0] = *(const i32x4*)(pb2 + k);
        B2.u[1] = *(const i32x4*)(pb2 + k + 32);
        B3.u[0] = *(const i32x4*)(pb3 + k);
        B3.u[1] = *(const i32x4*)(pb3 + k + 32);

        acc0 = __builtin_amdgcn_wmma_i32_16x16x64_iu8(true, A.v, true, B0.v, acc0, false, false);
        acc1 = __builtin_amdgcn_wmma_i32_16x16x64_iu8(true, A.v, true, B1.v, acc1, false, false);
        acc2 = __builtin_amdgcn_wmma_i32_16x16x64_iu8(true, A.v, true, B2.v, acc2, false, false);
        acc3 = __builtin_amdgcn_wmma_i32_16x16x64_iu8(true, A.v, true, B3.v, acc3, false, false);
    }

    i32x8 accs[4] = { acc0, acc1, acc2, acc3 };
    int   cols[4]; float wm[4], bs[4]; f32x4 wc[4][4];
#pragma unroll
    for (int t = 0; t < 4; ++t) {
        int col = n_base + t * 16 + lm;
        cols[t] = col; wm[t] = w_max[col]; bs[t] = bias[col];
        const f32x4* wp = (const f32x4*)(wcol + (size_t)col * K_OUT);
        wc[t][0] = wp[0]; wc[t][1] = wp[1]; wc[t][2] = wp[2]; wc[t][3] = wp[3];
    }
#pragma unroll
    for (int r = 0; r < 8; ++r) {
        int row = m_base + half * 8 + r;
        float xm = x_max[row];
        const f32x4* xp = (const f32x4*)(x_unq + (size_t)row * K_OUT);
        f32x4 x0 = xp[0], x1 = xp[1], x2 = xp[2], x3 = xp[3];
#pragma unroll
        for (int t = 0; t < 4; ++t) {
            f32x4 p = x0 * wc[t][0] + x1 * wc[t][1] + x2 * wc[t][2] + x3 * wc[t][3];
            float d = p.x + p.y + p.z + p.w;
            float qv = (float)(_Float16)((float)accs[t][r] / (QF * QF));
            out[(size_t)row * O_DIM + cols[t]] = qv * (xm * wm[t]) + d + bs[t];
        }
    }
}

extern "C" void kernel_launch(void* const* d_in, const int* in_sizes, int n_in,
                              void* d_out, int out_size, void* d_ws, size_t ws_size,
                              hipStream_t stream) {
    (void)in_sizes; (void)n_in; (void)out_size; (void)ws_size;
    const float* x    = (const float*)d_in[0];
    const float* w    = (const float*)d_in[1];
    const float* bias = (const float*)d_in[2];
    float* out = (float*)d_out;

    char* ws = (char*)d_ws;
    size_t off = 0;
    auto carve = [&](size_t bytes) -> void* {
        void* p = ws + off;
        off += (bytes + 255) & ~(size_t)255;
        return p;
    };
    float*       col_max = (float*)      carve(I_DIM * sizeof(float));
    int*         idx     = (int*)        carve(K_OUT * sizeof(int));
    float*       w_max   = (float*)      carve(O_DIM * sizeof(float));
    float*       x_maxp  = (float*)      carve(B_DIM * sizeof(float));
    float*       wcol    = (float*)      carve((size_t)O_DIM * K_OUT * 4);
    float*       x_unq   = (float*)      carve((size_t)B_DIM * K_OUT * 4);
    signed char* w_q     = (signed char*)carve((size_t)O_DIM * I_DIM);
    signed char* x_q     = (signed char*)carve((size_t)B_DIM * I_DIM);

    colmax_kernel<<<I_DIM / 256, 256, 0, stream>>>(w, col_max);
    topk_kernel<<<1, 256, 0, stream>>>(col_max, idx);
    quant_rows_kernel<<<O_DIM, 256, 0, stream>>>(w, idx, w_q, w_max, wcol);
    quant_rows_kernel<<<B_DIM, 256, 0, stream>>>(x, idx, x_q, x_maxp, x_unq);
    gemm_kernel<<<dim3(O_DIM / 64, B_DIM / 128), 256, 0, stream>>>(
        x_q, w_q, x_maxp, w_max, x_unq, wcol, bias, out);
}
